// GraphSAGEClassifier_60284160967394
// MI455X (gfx1250) — matrix operand order repaired and checked by reading, not compiled, not run on hardware
//
#include <hip/hip_runtime.h>

#define N_NODES  40000
#define N_EDGES  640000
#define DIM      128
#define OUT_DIM  16
#define N_GRAPHS 64

typedef __attribute__((ext_vector_type(16))) __bf16 v16bf;
typedef __attribute__((ext_vector_type(8)))  float  v8f;

union BfPack {
    v16bf v;
    unsigned short u[16];
    uint4 q[2];
};

// Native bf16 convert with degree scaling (lowers to v_pk_mul_f32 + v_cvt_pk_bf16_f32)
__device__ __forceinline__ void cvt16_scale(BfPack& p, const float4& a, const float4& b,
                                            const float4& c, const float4& d, float sc) {
    p.v[0]  = (__bf16)(a.x * sc); p.v[1]  = (__bf16)(a.y * sc);
    p.v[2]  = (__bf16)(a.z * sc); p.v[3]  = (__bf16)(a.w * sc);
    p.v[4]  = (__bf16)(b.x * sc); p.v[5]  = (__bf16)(b.y * sc);
    p.v[6]  = (__bf16)(b.z * sc); p.v[7]  = (__bf16)(b.w * sc);
    p.v[8]  = (__bf16)(c.x * sc); p.v[9]  = (__bf16)(c.y * sc);
    p.v[10] = (__bf16)(c.z * sc); p.v[11] = (__bf16)(c.w * sc);
    p.v[12] = (__bf16)(d.x * sc); p.v[13] = (__bf16)(d.y * sc);
    p.v[14] = (__bf16)(d.z * sc); p.v[15] = (__bf16)(d.w * sc);
}

// Unscaled variant: pure v_cvt_pk_bf16_f32, no multiplies
__device__ __forceinline__ void cvt16(BfPack& p, const float4& a, const float4& b,
                                      const float4& c, const float4& d) {
    p.v[0]  = (__bf16)a.x; p.v[1]  = (__bf16)a.y;
    p.v[2]  = (__bf16)a.z; p.v[3]  = (__bf16)a.w;
    p.v[4]  = (__bf16)b.x; p.v[5]  = (__bf16)b.y;
    p.v[6]  = (__bf16)b.z; p.v[7]  = (__bf16)b.w;
    p.v[8]  = (__bf16)c.x; p.v[9]  = (__bf16)c.y;
    p.v[10] = (__bf16)c.z; p.v[11] = (__bf16)c.w;
    p.v[12] = (__bf16)d.x; p.v[13] = (__bf16)d.y;
    p.v[14] = (__bf16)d.z; p.v[15] = (__bf16)d.w;
}

// ---- in-degree counts (same edge list for all 3 layers) ----
__global__ void degree_kernel(const long long* __restrict__ dst, float* __restrict__ cnt) {
    int e = blockIdx.x * blockDim.x + threadIdx.x;
    if (e < N_EDGES) atomicAdd(cnt + (int)dst[e], 1.0f);
}

// ---- f32 -> bf16 conversion of the six 128x128 SAGE weights ----
__global__ void pack_weights(const float* __restrict__ w0, const float* __restrict__ w1,
                             const float* __restrict__ w2, const float* __restrict__ w3,
                             const float* __restrict__ w4, const float* __restrict__ w5,
                             unsigned short* __restrict__ wpk) {
    int idx = blockIdx.x * blockDim.x + threadIdx.x;
    if (idx >= 6 * DIM * DIM) return;
    int wsel = idx / (DIM * DIM);
    int rem  = idx % (DIM * DIM);
    const float* W;
    switch (wsel) {
        case 0: W = w0; break; case 1: W = w1; break; case 2: W = w2; break;
        case 3: W = w3; break; case 4: W = w4; break; default: W = w5; break;
    }
    __bf16 h = (__bf16)W[rem];
    wpk[idx] = *(unsigned short*)&h;
}

// ---- scatter-add: one wave per edge, 4 channels per lane (coalesced row) ----
__global__ void scatter_kernel(const float* __restrict__ xin,
                               const long long* __restrict__ src,
                               const long long* __restrict__ dst,
                               float* __restrict__ s) {
    int tid = blockIdx.x * blockDim.x + threadIdx.x;   // N_EDGES*32 threads
    int e  = tid >> 5;
    int c4 = (tid & 31) * 4;
    int sn = (int)src[e];
    int dn = (int)dst[e];
    const float4 v = *(const float4*)(xin + sn * DIM + c4);
    float* p = s + dn * DIM + c4;
    atomicAdd(p + 0, v.x);
    atomicAdd(p + 1, v.y);
    atomicAdd(p + 2, v.z);
    atomicAdd(p + 3, v.w);
}

// ---- fused: (s/deg)@Wl^T + x@Wr^T + b, ReLU.  One wave = 16x16 tile, K=128 ----
__global__ __launch_bounds__(256) void sage_gemm_kernel(
    const float* __restrict__ xin, const float* __restrict__ s,
    const float* __restrict__ cnt,
    const unsigned short* __restrict__ wl, const unsigned short* __restrict__ wr,
    const float* __restrict__ bias, float* __restrict__ out) {

    const int lane = threadIdx.x & 31;
    const int wave = threadIdx.x >> 5;       // column tile 0..7  (H = 8*16)
    const int rowTile = blockIdx.x;          // 0..2499          (N = 2500*16)
    const int m  = lane & 15;
    const int hi = lane >> 4;

    const int arow = rowTile * 16 + m;       // A-matrix row this lane feeds
    const float inv = 1.0f / fmaxf(cnt[arow], 1.0f);

    const float* srow = s   + arow * DIM + hi * 16;  // A layout: 16 contiguous K
    const float* xrow = xin + arow * DIM + hi * 16;
    const int hcol = wave * 16 + m;                  // B column == C column
    const unsigned short* wlrow = wl + hcol * DIM + hi * 16;  // B layout: 16 contiguous K
    const unsigned short* wrrow = wr + hcol * DIM + hi * 16;

    v8f acc = {};
#pragma unroll
    for (int kt = 0; kt < 4; ++kt) {
        const int ko = kt * 32;
        BfPack aagg, ax, bl, br;
        // aggregated neighbors, normalized by degree
        cvt16_scale(aagg,
              *(const float4*)(srow + ko + 0),  *(const float4*)(srow + ko + 4),
              *(const float4*)(srow + ko + 8), *(const float4*)(srow + ko + 12), inv);
        // self features (no scale -> pure converts)
        cvt16(ax,
              *(const float4*)(xrow + ko + 0),  *(const float4*)(xrow + ko + 4),
              *(const float4*)(xrow + ko + 8), *(const float4*)(xrow + ko + 12));
        // bf16 weights: 32 contiguous bytes per lane
        const uint4* pl = (const uint4*)(wlrow + ko);
        const uint4* pr = (const uint4*)(wrrow + ko);
        bl.q[0] = pl[0]; bl.q[1] = pl[1];
        br.q[0] = pr[0]; br.q[1] = pr[1];

        acc = __builtin_amdgcn_wmma_f32_16x16x32_bf16(false, aagg.v, false, bl.v,
                                                      (short)0, acc, false, false);
        acc = __builtin_amdgcn_wmma_f32_16x16x32_bf16(false, ax.v,   false, br.v,
                                                      (short)0, acc, false, false);
    }

    const float bcol = bias[hcol];
#pragma unroll
    for (int r = 0; r < 8; ++r) {
        const int orow = rowTile * 16 + r + hi * 8;   // C/D layout
        const float v = acc[r] + bcol;
        out[orow * DIM + hcol] = fmaxf(v, 0.0f);      // ReLU
    }
}

// ---- global mean pool: segment-sum over sorted graph ids ----
__global__ void pool_kernel(const float* __restrict__ h, const long long* __restrict__ batch,
                            float* __restrict__ pooled, float* __restrict__ gcnt) {
    int tid = blockIdx.x * blockDim.x + threadIdx.x;  // N_NODES*32 threads
    int nd = tid >> 5;
    int c4 = (tid & 31) * 4;
    int g = (int)batch[nd];
    const float4 v = *(const float4*)(h + nd * DIM + c4);
    float* p = pooled + g * DIM + c4;
    atomicAdd(p + 0, v.x);
    atomicAdd(p + 1, v.y);
    atomicAdd(p + 2, v.z);
    atomicAdd(p + 3, v.w);
    if (c4 == 0) atomicAdd(gcnt + g, 1.0f);
}

// ---- final FC: [64,128] x [128,16] -- tiny, plain dot products ----
__global__ void fc_kernel(const float* __restrict__ pooled, const float* __restrict__ gcnt,
                          const float* __restrict__ Wfc, const float* __restrict__ bfc,
                          float* __restrict__ out) {
    int tid = blockIdx.x * blockDim.x + threadIdx.x;
    if (tid >= N_GRAPHS * OUT_DIM) return;
    int g = tid >> 4;
    int o = tid & 15;
    const float inv = 1.0f / fmaxf(gcnt[g], 1.0f);
    const float* pr = pooled + g * DIM;
    const float* wr = Wfc + o * DIM;
    float sum = 0.0f;
#pragma unroll 4
    for (int k = 0; k < DIM; ++k) sum += pr[k] * wr[k];
    out[tid] = bfc[o] + sum * inv;
}

extern "C" void kernel_launch(void* const* d_in, const int* in_sizes, int n_in,
                              void* d_out, int out_size, void* d_ws, size_t ws_size,
                              hipStream_t stream) {
    const float*     x     = (const float*)d_in[0];
    const long long* eidx  = (const long long*)d_in[1];
    const long long* batch = (const long long*)d_in[2];
    const float* W1l = (const float*)d_in[3];
    const float* W1r = (const float*)d_in[4];
    const float* b1  = (const float*)d_in[5];
    const float* W2l = (const float*)d_in[6];
    const float* W2r = (const float*)d_in[7];
    const float* b2  = (const float*)d_in[8];
    const float* W3l = (const float*)d_in[9];
    const float* W3r = (const float*)d_in[10];
    const float* b3  = (const float*)d_in[11];
    const float* Wfc = (const float*)d_in[12];
    const float* bfc = (const float*)d_in[13];

    const long long* src = eidx;
    const long long* dst = eidx + N_EDGES;

    // workspace layout (all offsets 256B-aligned)
    char* ws = (char*)d_ws;
    const size_t SZ_H = (size_t)N_NODES * DIM * sizeof(float);   // 20,480,000
    float* S    = (float*)(ws);
    float* hA   = (float*)(ws + SZ_H);
    float* hB   = (float*)(ws + 2 * SZ_H);
    float* cnt  = (float*)(ws + 3 * SZ_H);                       // 160,000 B
    unsigned short* wpk = (unsigned short*)(ws + 3 * SZ_H + 160000);  // 196,608 B
    float* pooled = (float*)(ws + 3 * SZ_H + 160000 + 196608);   // 32,768 B
    float* gcnt   = (float*)(ws + 3 * SZ_H + 160000 + 196608 + 32768);

    // one-time-per-call prep (deterministic)
    hipMemsetAsync(cnt, 0, N_NODES * sizeof(float), stream);
    degree_kernel<<<(N_EDGES + 255) / 256, 256, 0, stream>>>(dst, cnt);
    pack_weights<<<(6 * DIM * DIM + 255) / 256, 256, 0, stream>>>(
        W1l, W1r, W2l, W2r, W3l, W3r, wpk);

    const float* lin[3]  = { x,  hA, hB };
    float*       lout[3] = { hA, hB, hA };
    const float* lb[3]   = { b1, b2, b3 };

    for (int L = 0; L < 3; ++L) {
        hipMemsetAsync(S, 0, SZ_H, stream);
        scatter_kernel<<<(N_EDGES * 32) / 256, 256, 0, stream>>>(lin[L], src, dst, S);
        sage_gemm_kernel<<<N_NODES / 16, 256, 0, stream>>>(
            lin[L], S, cnt,
            wpk + (2 * L + 0) * (DIM * DIM),
            wpk + (2 * L + 1) * (DIM * DIM),
            lb[L], lout[L]);
    }

    hipMemsetAsync(pooled, 0, N_GRAPHS * DIM * sizeof(float), stream);
    hipMemsetAsync(gcnt,   0, N_GRAPHS * sizeof(float), stream);
    pool_kernel<<<(N_NODES * 32) / 256, 256, 0, stream>>>(hA, batch, pooled, gcnt);
    fc_kernel<<<(N_GRAPHS * OUT_DIM + 255) / 256, 256, 0, stream>>>(
        pooled, gcnt, Wfc, bfc, (float*)d_out);
}
